// LSHAttention_163208757699
// MI455X (gfx1250) — matrix operand order repaired and checked by reading, not compiled, not run on hardware
//
#include <hip/hip_runtime.h>
#include <hip/hip_bf16.h>
#include <math.h>

// ---------------- problem constants (fixed by setup_inputs) ----------------
#define B_    8
#define SL_   4096
#define D_    128
#define NH_   8
#define NB_   64              // n_buckets
#define NC_   512             // n_chunks per batch (NH_ * NB_)
#define NHSL_ (NH_ * SL_)     // 32768
#define BS_   64              // BUCKET_SIZE

typedef __attribute__((ext_vector_type(16))) __bf16 v16bf;
typedef __attribute__((ext_vector_type(8)))  float  v8f;

// ============================================================================
// Kernel 1: LSH hashing. One block per token; thread t = (h, n) pair.
//   rotated[h][n] = sum_d qk[b,s,d] * rot[d,h,n]; bucket = argmax over [rot,-rot]
// ============================================================================
__global__ __launch_bounds__(256) void lsh_hash_kernel(
    const float* __restrict__ qk, const float* __restrict__ rot,
    int* __restrict__ buckets)
{
    const int tok = blockIdx.x;                 // 0 .. B_*SL_-1
    const int bb  = tok / SL_;
    const int s   = tok % SL_;
    const int tid = threadIdx.x;

    __shared__ float qrow[D_];
    if (tid < D_) qrow[tid] = qk[((size_t)bb * SL_ + s) * D_ + tid];
    __syncthreads();

    const int h = tid >> 5;                     // 0..7
    const int n = tid & 31;                     // 0..31
    float acc = 0.0f;
    #pragma unroll 8
    for (int dd = 0; dd < D_; ++dd)
        acc += qrow[dd] * rot[dd * 256 + tid];  // rot[d][h][n], h*32+n == tid

    // local argmax over {+acc at n, -acc at n+32}; first-index tie-break
    float bv; int bi;
    if (acc >= -acc) { bv = acc;  bi = n; }
    else             { bv = -acc; bi = n + 32; }

    // wave32 butterfly argmax (first index wins ties)
    #pragma unroll
    for (int off = 16; off > 0; off >>= 1) {
        float ov = __shfl_xor(bv, off);
        int   oi = __shfl_xor(bi, off);
        if (ov > bv || (ov == bv && oi < bi)) { bv = ov; bi = oi; }
    }
    if ((tid & 31) == 0)
        buckets[(size_t)bb * NHSL_ + h * SL_ + s] = bi + h * NB_;
}

// ============================================================================
// Kernel 2: stable counting sort per (b, h) segment (keys disjoint per h).
// Emits s_pos (sorted source positions), s_bucket (sorted global buckets),
// undo (original index -> sorted index).
// ============================================================================
__global__ __launch_bounds__(256) void lsh_sort_kernel(
    const int* __restrict__ buckets,
    int* __restrict__ s_pos, int* __restrict__ s_bucket, int* __restrict__ undo)
{
    const int bb = blockIdx.x >> 3;
    const int h  = blockIdx.x & 7;
    const int tid = threadIdx.x;

    __shared__ int lb[SL_];
    __shared__ int cnt[NB_];
    __shared__ int basep[NB_];

    if (tid < NB_) cnt[tid] = 0;
    __syncthreads();

    for (int s = tid; s < SL_; s += 256) {
        int g = buckets[(size_t)bb * NHSL_ + h * SL_ + s] - h * NB_;
        lb[s] = g;
        atomicAdd(&cnt[g], 1);
    }
    __syncthreads();

    if (tid == 0) {
        int acc = 0;
        for (int j = 0; j < NB_; ++j) { basep[j] = acc; acc += cnt[j]; }
    }
    __syncthreads();

    if (tid < NB_) {                 // thread j scans positions in order -> stable
        const int j = tid;
        int cur = basep[j];
        for (int s = 0; s < SL_; ++s) {
            if (lb[s] == j) {
                int ig = h * SL_ + cur;
                s_pos   [(size_t)bb * NHSL_ + ig] = s;
                s_bucket[(size_t)bb * NHSL_ + ig] = j + h * NB_;
                undo    [(size_t)bb * NHSL_ + h * SL_ + s] = ig;
                ++cur;
            }
        }
    }
}

// ============================================================================
// Kernel 3: duplicate-attention locs table: locs[b][s][h]   = bkt*NC + chunk
//                                           locs[b][s][8+h] = bkt*NC + (chunk+1)%NC
// ============================================================================
__global__ __launch_bounds__(256) void lsh_locs_kernel(
    const int* __restrict__ buckets, const int* __restrict__ undo,
    int* __restrict__ locsbuf)
{
    int t  = blockIdx.x * 256 + threadIdx.x;    // B_*SL_*NH_ threads
    int h  = t & 7;
    int s  = (t >> 3) & (SL_ - 1);
    int bb = t >> 15;
    int j  = h * SL_ + s;
    int bkt = buckets[(size_t)bb * NHSL_ + j];
    int ch  = undo  [(size_t)bb * NHSL_ + j] >> 6;   // /BUCKET_SIZE
    size_t base = ((size_t)bb * SL_ + s) * 16;
    locsbuf[base + h]     = bkt * NC_ + ch;
    locsbuf[base + 8 + h] = bkt * NC_ + ((ch + 1) & (NC_ - 1));
}

// ============================================================================
// Kernel 4: per-chunk LSH attention. 1 block (8 waves) per (b, chunk).
// bf16 WMMA (v_wmma_f32_16x16x32_bf16) for QK^T and PV; f32 masks/softmax.
// ============================================================================
__global__ __launch_bounds__(256) void lsh_attn_chunk_kernel(
    const float* __restrict__ qk, const float* __restrict__ vin,
    const int* __restrict__ s_pos, const int* __restrict__ s_bucket,
    const int* __restrict__ locsbuf,
    float* __restrict__ so, float* __restrict__ slogits)
{
    const int c  = blockIdx.x % NC_;
    const int bb = blockIdx.x / NC_;
    const int tid  = threadIdx.x;
    const int wave = tid >> 5;       // 0..7 -> owns N-tile `wave`
    const int lane = tid & 31;
    const int hi   = lane >> 4;      // lane half (WMMA fragment layout)
    const int ln   = lane & 15;

    __shared__ __bf16 sQP[64 * 128];   // Q (bf16), later reused as P
    __shared__ __bf16 sK [128 * 128];  // normalized keys
    __shared__ __bf16 sV [128 * 128];
    __shared__ float  sD [64 * 128];   // dots
    __shared__ int    s_tq[64], s_bq[64], s_tkv[128], s_bkv[128];
    __shared__ int    s_qlocs[64 * 8], s_kvlocs[128 * 16];
    __shared__ float  s_kinv[128];

    // ---- metadata ----
    if (tid < 64) {
        int ig = c * BS_ + tid;
        s_tq[tid] = s_pos   [(size_t)bb * NHSL_ + ig];
        s_bq[tid] = s_bucket[(size_t)bb * NHSL_ + ig];
    }
    if (tid < 128) {
        int j  = tid;
        int cc = (j < 64) ? c : (c + NC_ - 1) % NC_;   // look_one_back
        int ig = cc * BS_ + (j & 63);
        s_tkv[j] = s_pos   [(size_t)bb * NHSL_ + ig];
        s_bkv[j] = s_bucket[(size_t)bb * NHSL_ + ig];
    }
    __syncthreads();

    // ---- key norms, locs, Q and V tiles ----
    if (tid < 128) {
        const float* row = qk + ((size_t)bb * SL_ + s_tkv[tid]) * D_;
        float ss = 0.0f;
        #pragma unroll 8
        for (int dd = 0; dd < D_; ++dd) ss += row[dd] * row[dd];
        s_kinv[tid] = 1.0f / fmaxf(sqrtf(ss), 1e-12f);
    }
    for (int e = tid; e < 64 * 8; e += 256) {
        int i = e >> 3, c8 = e & 7;
        s_qlocs[e] = locsbuf[((size_t)bb * SL_ + s_tq[i]) * 16 + c8];
    }
    for (int e = tid; e < 128 * 16; e += 256) {
        int j = e >> 4, c16 = e & 15;
        s_kvlocs[e] = locsbuf[((size_t)bb * SL_ + s_tkv[j]) * 16 + c16];
    }
    for (int e = tid; e < 64 * 128; e += 256) {
        int i = e >> 7, dd = e & 127;
        sQP[e] = (__bf16)qk[((size_t)bb * SL_ + s_tq[i]) * D_ + dd];
    }
    for (int e = tid; e < 128 * 128; e += 256) {
        int j = e >> 7, dd = e & 127;
        sV[e] = (__bf16)vin[((size_t)bb * SL_ + s_tkv[j]) * D_ + dd];
    }
    __syncthreads();
    for (int e = tid; e < 128 * 128; e += 256) {
        int j = e >> 7, dd = e & 127;
        sK[e] = (__bf16)(qk[((size_t)bb * SL_ + s_tkv[j]) * D_ + dd] * s_kinv[j]);
    }
    __syncthreads();

    // ---- dots = Q @ Knorm^T : wave w -> N-tile w (kv cols), 4 M-tiles, K=128/32
    {
        const v8f zero = {0.f,0.f,0.f,0.f,0.f,0.f,0.f,0.f};
        v8f acc[4] = {zero, zero, zero, zero};
        for (int ks = 0; ks < 4; ++ks) {
            v16bf bfrag;                       // B(32xk,16n): lanes 0-15 K0..15
            const int kb = ks * 32 + hi * 16;
            #pragma unroll
            for (int s2 = 0; s2 < 16; ++s2)
                bfrag[s2] = sK[(wave * 16 + ln) * 128 + kb + s2];
            #pragma unroll
            for (int mt = 0; mt < 4; ++mt) {
                v16bf afrag;                   // A(16m,32k): slots 0-15 K b..b+15
                const int ab = ks * 32 + hi * 16;
                const int m  = mt * 16 + ln;
                #pragma unroll
                for (int s2 = 0; s2 < 8; ++s2)  afrag[s2] = sQP[m * 128 + ab + s2];
                #pragma unroll
                for (int s2 = 8; s2 < 16; ++s2) afrag[s2] = sQP[m * 128 + ab + s2];
                acc[mt] = __builtin_amdgcn_wmma_f32_16x16x32_bf16(
                    false, afrag, false, bfrag, (short)0, acc[mt], false, false);
            }
        }
        #pragma unroll
        for (int mt = 0; mt < 4; ++mt)
            #pragma unroll
            for (int r = 0; r < 8; ++r)
                sD[(mt * 16 + r + hi * 8) * 128 + wave * 16 + ln] = acc[mt][r];
    }
    __syncthreads();

    // ---- scale + masks + duplicate-count correction ----
    const float scale = 0.08838834764831845f;          // 128^-0.5
    const float MNEG  = -3.402823466e38f;
    for (int e = tid; e < 64 * 128; e += 256) {
        int i = e >> 7, j = e & 127;
        float x = sD[e] * scale;
        int tq = s_tq[i], tk = s_tkv[j];
        if (tq <  tk) x = MNEG;                        // causal
        if (tq == tk) x = -10000.0f;                   // self-attn penalty
        if (s_bq[i] != s_bkv[j]) x = MNEG;             // cross-bucket
        int cnt = 0;
        #pragma unroll
        for (int c8 = 0; c8 < 8; ++c8) {
            int q = s_qlocs[i * 8 + c8];
            cnt += (q == s_kvlocs[j * 16 + c8]);
            cnt += (q == s_kvlocs[j * 16 + 8 + c8]);
        }
        sD[e] = x - logf((float)cnt + 1e-9f);
    }
    __syncthreads();

    // ---- row logsumexp + P = exp(x - lse)  (wave handles 8 rows) ----
    for (int rr = 0; rr < 8; ++rr) {
        int row = wave * 8 + rr;
        float x0 = sD[row * 128 + lane];
        float x1 = sD[row * 128 + lane + 32];
        float x2 = sD[row * 128 + lane + 64];
        float x3 = sD[row * 128 + lane + 96];
        float m = fmaxf(fmaxf(x0, x1), fmaxf(x2, x3));
        #pragma unroll
        for (int off = 16; off > 0; off >>= 1) m = fmaxf(m, __shfl_xor(m, off));
        float sm = expf(x0 - m) + expf(x1 - m) + expf(x2 - m) + expf(x3 - m);
        #pragma unroll
        for (int off = 16; off > 0; off >>= 1) sm += __shfl_xor(sm, off);
        float lse = m + logf(sm);
        if (lane == 0)
            slogits[(size_t)bb * NHSL_ + c * BS_ + row] = lse;
        sQP[row * 128 + lane]      = (__bf16)expf(x0 - lse);
        sQP[row * 128 + lane + 32] = (__bf16)expf(x1 - lse);
        sQP[row * 128 + lane + 64] = (__bf16)expf(x2 - lse);
        sQP[row * 128 + lane + 96] = (__bf16)expf(x3 - lse);
    }
    __syncthreads();

    // ---- out = P @ V : wave w -> d-cols tile w ----
    {
        const v8f zero = {0.f,0.f,0.f,0.f,0.f,0.f,0.f,0.f};
        v8f acc[4] = {zero, zero, zero, zero};
        for (int ks = 0; ks < 4; ++ks) {
            v16bf bfrag;                       // B: K = kv rows, N = d cols
            const int kb = ks * 32 + hi * 16;
            #pragma unroll
            for (int s2 = 0; s2 < 16; ++s2)
                bfrag[s2] = sV[(kb + s2) * 128 + wave * 16 + ln];
            #pragma unroll
            for (int mt = 0; mt < 4; ++mt) {
                v16bf afrag;                   // A: rows = q, K = kv
                const int ab = ks * 32 + hi * 16;
                const int m  = mt * 16 + ln;
                #pragma unroll
                for (int s2 = 0; s2 < 8; ++s2)  afrag[s2] = sQP[m * 128 + ab + s2];
                #pragma unroll
                for (int s2 = 8; s2 < 16; ++s2) afrag[s2] = sQP[m * 128 + ab + s2];
                acc[mt] = __builtin_amdgcn_wmma_f32_16x16x32_bf16(
                    false, afrag, false, bfrag, (short)0, acc[mt], false, false);
            }
        }
        #pragma unroll
        for (int mt = 0; mt < 4; ++mt)
            #pragma unroll
            for (int r = 0; r < 8; ++r) {
                int m   = mt * 16 + r + hi * 8;
                int col = wave * 16 + ln;
                so[((size_t)bb * NHSL_ + c * BS_ + m) * D_ + col] = acc[mt][r];
            }
    }
}

// ============================================================================
// Kernel 5: unsort + combine across hash rounds with softmax over LSEs.
// ============================================================================
__global__ __launch_bounds__(128) void lsh_combine_kernel(
    const int* __restrict__ undo, const float* __restrict__ slogits,
    const float* __restrict__ so, float* __restrict__ out)
{
    const int bb = blockIdx.x / SL_;
    const int s  = blockIdx.x % SL_;
    const int d  = threadIdx.x;

    __shared__ int   idx8[NH_];
    __shared__ float ls8[NH_];
    if (threadIdx.x < NH_) {
        int h  = threadIdx.x;
        int ii = undo[(size_t)bb * NHSL_ + h * SL_ + s];
        idx8[h] = ii;
        ls8[h]  = slogits[(size_t)bb * NHSL_ + ii];
    }
    __syncthreads();

    float m = -3.402823466e38f;
    #pragma unroll
    for (int h = 0; h < NH_; ++h) m = fmaxf(m, ls8[h]);
    float sm = 0.0f;
    #pragma unroll
    for (int h = 0; h < NH_; ++h) sm += expf(ls8[h] - m);
    float acc = 0.0f;
    #pragma unroll
    for (int h = 0; h < NH_; ++h)
        acc += (expf(ls8[h] - m) / sm) *
               so[((size_t)bb * NHSL_ + idx8[h]) * D_ + d];
    out[((size_t)bb * SL_ + s) * D_ + d] = acc;
}

// ============================================================================
extern "C" void kernel_launch(void* const* d_in, const int* in_sizes, int n_in,
                              void* d_out, int out_size, void* d_ws, size_t ws_size,
                              hipStream_t stream)
{
    const float* qk  = (const float*)d_in[0];   // [B, SL, D]
    const float* v   = (const float*)d_in[1];   // [B, SL, D]
    const float* rot = (const float*)d_in[2];   // [D, NH, NB/2]
    float* out = (float*)d_out;                 // [B, SL, D]

    // workspace layout (~141 MB total)
    char* ws = (char*)d_ws;
    size_t off = 0;
    auto take = [&](size_t bytes) -> char* {
        char* p = ws + off;
        off = (off + bytes + 255) & ~(size_t)255;
        return p;
    };
    int*   buckets  = (int*)  take((size_t)B_ * NHSL_ * sizeof(int));
    int*   s_pos    = (int*)  take((size_t)B_ * NHSL_ * sizeof(int));
    int*   s_bucket = (int*)  take((size_t)B_ * NHSL_ * sizeof(int));
    int*   undo     = (int*)  take((size_t)B_ * NHSL_ * sizeof(int));
    int*   locsbuf  = (int*)  take((size_t)B_ * SL_ * 16 * sizeof(int));
    float* slogits  = (float*)take((size_t)B_ * NHSL_ * sizeof(float));
    float* so       = (float*)take((size_t)B_ * NHSL_ * D_ * sizeof(float));

    lsh_hash_kernel<<<B_ * SL_, 256, 0, stream>>>(qk, rot, buckets);
    lsh_sort_kernel<<<B_ * NH_, 256, 0, stream>>>(buckets, s_pos, s_bucket, undo);
    lsh_locs_kernel<<<(B_ * SL_ * NH_) / 256, 256, 0, stream>>>(buckets, undo, locsbuf);
    lsh_attn_chunk_kernel<<<B_ * NC_, 256, 0, stream>>>(qk, v, s_pos, s_bucket,
                                                        locsbuf, so, slogits);
    lsh_combine_kernel<<<B_ * SL_, 128, 0, stream>>>(undo, slogits, so, out);
}
